// LSTM_78855599554848
// MI455X (gfx1250) — compile-verified, matrix operand order checked by reading, not run on hardware
//
#include <hip/hip_runtime.h>
#include <stdint.h>

// ---------------------------------------------------------------------------
// LSTM on MI455X (gfx1250): bf16 WMMA (v_wmma_f32_16x16x32_bf16) with f32
// accumulation; async global->LDS staging (ASYNCcnt) for the projection GEMM.
// Gate order i, f, g, o along the leading axis of Wx/Wh.
// ---------------------------------------------------------------------------

typedef __attribute__((ext_vector_type(16))) __bf16 v16bf;
typedef __attribute__((ext_vector_type(8)))  __bf16 v8bf;
typedef __attribute__((ext_vector_type(8)))  float  v8f;

#define S_LEN 512
#define BATCH 64
#define IDIM  1024
#define HDIM  1024

__device__ __forceinline__ unsigned short f32_bf16(float f) {
  union { float f; unsigned u; } v; v.f = f;
  unsigned u = v.u;
  unsigned r = 0x7FFFu + ((u >> 16) & 1u);   // round-to-nearest-even
  return (unsigned short)((u + r) >> 16);
}

// Build a 16-element bf16 fragment from two contiguous 8-element (16-byte)
// chunks. Per the CDNA5 16-bit A/B layout (16x32 tile, row-major ld=K):
//   lane<16 : K = {0..7} U {16..23};  lane>=16: K = {8..15} U {24..31}
__device__ __forceinline__ v16bf load_frag(const unsigned short* p) {
  v8bf lo = *(const v8bf*)(p);
  v8bf hi = *(const v8bf*)(p + 16);
  return __builtin_shufflevector(lo, hi, 0,1,2,3,4,5,6,7,8,9,10,11,12,13,14,15);
}

__device__ __forceinline__ float sigmoid_f(float x) {
  return 1.0f / (1.0f + __expf(-x));
}
__device__ __forceinline__ float tanh_f(float x) {
  float e = __expf(2.0f * x);
  return 1.0f - 2.0f / (e + 1.0f);     // saturates cleanly for |x| large
}

// CDNA5 async copy: global -> LDS, 16 bytes per lane, tracked by ASYNCcnt.
// VDST = VGPR holding the LDS byte address (generic addr low 32 bits).
#define ASYNC_G2L_B128(ldsoff, gptr)                                        \
  asm volatile("global_load_async_to_lds_b128 %0, %1, off"                  \
               :: "v"(ldsoff),                                              \
                  "v"((unsigned long long)(uintptr_t)(gptr)) : "memory")

#define WAIT_ASYNC_3() asm volatile("s_wait_asynccnt 3" ::: "memory")

// ---------------------------------------------------------------------------
// Prep kernels
// ---------------------------------------------------------------------------
__global__ void convert_bf16_kernel(const float* __restrict__ in,
                                    unsigned short* __restrict__ out, int n4) {
  int i = blockIdx.x * 256 + threadIdx.x;
  if (i < n4) {
    float4 f = ((const float4*)in)[i];
    ushort4 o;
    o.x = f32_bf16(f.x); o.y = f32_bf16(f.y);
    o.z = f32_bf16(f.z); o.w = f32_bf16(f.w);
    ((ushort4*)out)[i] = o;
  }
}

// W[g][k][n] (f32) -> WT[g][n][k] (bf16), g in [0,4), k,n in [0,1024)
__global__ void transpose_w_kernel(const float* __restrict__ W,
                                   unsigned short* __restrict__ WT) {
  int i = blockIdx.x * 256 + threadIdx.x;          // index into WT
  int k = i & 1023, n = (i >> 10) & 1023, g = i >> 20;
  WT[i] = f32_bf16(W[(((size_t)g * 1024 + k) * 1024) + n]);
}

__global__ void zero_state_kernel(float* __restrict__ c,
                                  unsigned short* __restrict__ hb) {
  int i = blockIdx.x * 256 + threadIdx.x;          // 64*1024 = 65536 elements
  c[i]  = 0.0f;
  hb[i] = 0;
}

// ---------------------------------------------------------------------------
// Phase 1: xp[s][g][b][h] = x[b][s][:] @ Wx[g][:,h] + bx[g][h] + bh[g][h]
// M = B*S = 32768 (row r = b*512 + s), N = 4*H = 4096, K = 1024.
// Block: 256 threads = 8 waves; macro tile 64(M) x 128(N); each wave owns a
// 16x64 strip (4 WMMAs per 32-deep K-step). LDS double-buffered; tiles are
// staged with global_load_async_to_lds_b128 and overlapped with compute.
// ---------------------------------------------------------------------------
__global__ __launch_bounds__(256) void gemm_xp_kernel(
    const unsigned short* __restrict__ xb,    // [32768][1024] bf16
    const unsigned short* __restrict__ WxT,   // [4096][1024]  bf16 (N-major)
    const float* __restrict__ bx,             // [4096]
    const float* __restrict__ bh,             // [4096]
    float* __restrict__ xp)                   // [512][4][64][1024]
{
  __shared__ unsigned short As[2][64 * 32];    // 2 x 4 KB
  __shared__ unsigned short Bs[2][128 * 32];   // 2 x 8 KB

  const int t    = threadIdx.x;
  const int lane = t & 31;
  const int w    = t >> 5;
  const int wm   = w & 3;           // wave M-tile (0..3)
  const int wq   = w >> 2;          // wave N-group (0..1), 64 cols each
  const int bm   = blockIdx.x * 64;
  const int bn   = blockIdx.y * 128;
  const int mlo  = lane & 15;
  const int kh   = (lane >> 4) << 3;

  // per-thread staging slots (16 B each)
  const int aIdx = t << 3;                       // A: 64x32 = 2048 elems
  const int aRow = aIdx >> 5, aCol = aIdx & 31;
  const int bRow = aIdx >> 5, bCol = aIdx & 31;  // B chunk0; chunk1 = +64 rows

  const unsigned short* gA  = xb  + (size_t)(bm + aRow) * IDIM + aCol;
  const unsigned short* gB0 = WxT + (size_t)(bn + bRow) * IDIM + bCol;
  const unsigned short* gB1 = gB0 + (size_t)64 * IDIM;

  unsigned ldsA[2], ldsB0[2], ldsB1[2];
#pragma unroll
  for (int p = 0; p < 2; ++p) {
    ldsA[p]  = (unsigned)(uintptr_t)&As[p][aIdx];
    ldsB0[p] = (unsigned)(uintptr_t)&Bs[p][aIdx];
    ldsB1[p] = (unsigned)(uintptr_t)&Bs[p][aIdx + 2048];
  }

  v8f acc[4] = {};

  // prologue: stage k-tile 0 into buffer 0
  ASYNC_G2L_B128(ldsA[0],  gA);
  ASYNC_G2L_B128(ldsB0[0], gB0);
  ASYNC_G2L_B128(ldsB1[0], gB1);

  for (int i = 0; i < 32; ++i) {
    const int cur = i & 1, nxt = cur ^ 1;
    const int kn = (i < 31) ? (i + 1) * 32 : 0;     // wrap: harmless refetch
    // prefetch next k-tile (buffer was drained at previous end-barrier)
    ASYNC_G2L_B128(ldsA[nxt],  gA  + kn);
    ASYNC_G2L_B128(ldsB0[nxt], gB0 + kn);
    ASYNC_G2L_B128(ldsB1[nxt], gB1 + kn);
    // wait until only the 3 just-issued remain outstanding -> cur complete
    WAIT_ASYNC_3();
    __syncthreads();

    v16bf a = load_frag(&As[cur][(wm * 16 + mlo) * 32 + kh]);
#pragma unroll
    for (int nt = 0; nt < 4; ++nt) {
      v16bf b = load_frag(&Bs[cur][(wq * 64 + nt * 16 + mlo) * 32 + kh]);
      acc[nt] = __builtin_amdgcn_wmma_f32_16x16x32_bf16(
          false, a, false, b, (short)0, acc[nt], false, false);
    }
    __syncthreads();   // drain before this buffer is refilled next iteration
  }

  // Epilogue: D layout — lane holds col n = lane&15, rows m = r + 8*(lane>>4)
  const int mrow = bm + wm * 16 + ((lane >> 4) << 3);
#pragma unroll
  for (int nt = 0; nt < 4; ++nt) {
    const int ng = bn + wq * 64 + nt * 16 + mlo;   // global N index
    const float bs = bx[ng] + bh[ng];
    const int g = ng >> 10;
    const int h = ng & 1023;
#pragma unroll
    for (int r = 0; r < 8; ++r) {
      int R = mrow + r;
      int b = R >> 9;                              // batch
      int s = R & 511;                             // time step
      xp[((((size_t)s * 4 + g) * 64) + b) * 1024 + h] = acc[nt][r] + bs;
    }
  }
}

// ---------------------------------------------------------------------------
// Phase 2: one time step, fully fused.
// gates[g] = h @ WhT[g] + xp_t[g];  c' = sig(f)*c + tanh(g)*sig(i);
// h' = sig(o)*tanh(c').  Grid: 256 blocks (4 batch-tiles x 64 H-blocks) x
// 128 threads (wave == gate). Wh (8.4 MB bf16) and h (128 KB) are L2-resident.
// ---------------------------------------------------------------------------
__global__ __launch_bounds__(128) void lstm_step_kernel(
    const unsigned short* __restrict__ hb_in,   // [64][1024] bf16
    unsigned short* __restrict__ hb_out,        // [64][1024] bf16
    const unsigned short* __restrict__ WhT,     // [4][1024(n)][1024(k)] bf16
    const float* __restrict__ xp_s,             // [4][64][1024]
    float* __restrict__ c,                      // [64][1024] (in place)
    float* __restrict__ out,                    // d_out: h then c (f32)
    int is_last)
{
  __shared__ float gbuf[4][16][16];

  const int t    = threadIdx.x;
  const int lane = t & 31;
  const int g    = t >> 5;                 // gate handled by this wave
  const int mt   = blockIdx.x & 3;         // batch tile (16 rows)
  const int bn   = (blockIdx.x >> 2) * 16; // H-column block
  const int nl   = lane & 15;
  const int kh   = (lane >> 4) << 3;

  v8f acc = {};
  const unsigned short* wrow = WhT + ((size_t)g * 1024 + bn + nl) * HDIM;
  const unsigned short* arow = hb_in + (size_t)(mt * 16 + nl) * HDIM;

  for (int kk = 0; kk < HDIM; kk += 32) {
    v16bf bf = load_frag(wrow + kk + kh);
    v16bf af = load_frag(arow + kk + kh);
    acc = __builtin_amdgcn_wmma_f32_16x16x32_bf16(
        false, af, false, bf, (short)0, acc, false, false);
  }

  // gate tile -> LDS (+ input projection)
  const int mhi = (lane >> 4) << 3;
#pragma unroll
  for (int r = 0; r < 8; ++r) {
    int m = mhi + r;                               // 0..15 within tile
    gbuf[g][m][nl] = acc[r] +
                     xp_s[((size_t)g * 64 + mt * 16 + m) * 1024 + bn + nl];
  }
  __syncthreads();

  // pointwise LSTM cell update; each element owned by exactly one thread
#pragma unroll
  for (int k = 0; k < 2; ++k) {
    int j   = k * 128 + t;                 // 0..255 over (b_local, col)
    int bl  = j >> 4;
    int col = j & 15;
    float ig = sigmoid_f(gbuf[0][bl][col]);
    float fg = sigmoid_f(gbuf[1][bl][col]);
    float gg = tanh_f(gbuf[2][bl][col]);
    float og = sigmoid_f(gbuf[3][bl][col]);
    size_t ci = (size_t)(mt * 16 + bl) * 1024 + bn + col;
    float cn  = c[ci] * fg + gg * ig;
    float hn  = og * tanh_f(cn);
    c[ci]      = cn;
    hb_out[ci] = f32_bf16(hn);
    if (is_last) {
      out[ci]                 = hn;        // h
      out[(size_t)65536 + ci] = cn;        // c
    }
  }
}

// ---------------------------------------------------------------------------
// Host-side orchestration
// ---------------------------------------------------------------------------
extern "C" void kernel_launch(void* const* d_in, const int* in_sizes, int n_in,
                              void* d_out, int out_size, void* d_ws, size_t ws_size,
                              hipStream_t stream) {
  (void)in_sizes; (void)n_in; (void)out_size; (void)ws_size;
  const float* x  = (const float*)d_in[0];   // [64][512][1024]
  const float* Wx = (const float*)d_in[1];   // [4][1024][1024]
  const float* Wh = (const float*)d_in[2];   // [4][1024][1024]
  const float* bx = (const float*)d_in[3];   // [4][1024]
  const float* bh = (const float*)d_in[4];   // [4][1024]
  float* out = (float*)d_out;

  // workspace layout (bytes)
  char* ws = (char*)d_ws;
  unsigned short* xb  = (unsigned short*)(ws);                    // 67108864
  unsigned short* WxT = (unsigned short*)(ws + 67108864);         //  8388608
  unsigned short* WhT = (unsigned short*)(ws + 75497472);         //  8388608
  unsigned short* hb0 = (unsigned short*)(ws + 83886080);         //   131072
  unsigned short* hb1 = (unsigned short*)(ws + 84017152);         //   131072
  float*          c   = (float*)(ws + 84148224);                  //   262144
  float*          xp  = (float*)(ws + 84410368);                  // 536870912

  // prep: bf16 conversions + weight transposes + zero state
  convert_bf16_kernel<<<32768, 256, 0, stream>>>(x, xb, (64 * 512 * 1024) / 4);
  transpose_w_kernel<<<16384, 256, 0, stream>>>(Wx, WxT);
  transpose_w_kernel<<<16384, 256, 0, stream>>>(Wh, WhT);
  zero_state_kernel<<<256, 256, 0, stream>>>(c, hb0);

  // phase 1: all input projections in one GEMM (M=32768, N=4096, K=1024)
  gemm_xp_kernel<<<dim3(512, 32), 256, 0, stream>>>(xb, WxT, bx, bh, xp);

  // phase 2: sequential recurrence (ping-pong bf16 h buffers)
  unsigned short* hbuf[2] = { hb0, hb1 };
  for (int s = 0; s < S_LEN; ++s) {
    lstm_step_kernel<<<256, 128, 0, stream>>>(
        hbuf[s & 1], hbuf[(s + 1) & 1], WhT,
        xp + (size_t)s * 4 * 64 * 1024, c, out, (s == S_LEN - 1) ? 1 : 0);
  }
}
